// LocalDynamics_6571299963003
// MI455X (gfx1250) — compile-verified, matrix operand order checked by reading, not run on hardware
//
#include <hip/hip_runtime.h>
#include <hip/hip_bf16.h>
#include <math.h>

// ---------------------------------------------------------------------------
// LocalDynamics message-passing step for gfx1250 (MI455X), wave32 + WMMA.
//   delta = scatter_add over edges of tanh(MLP(edge_input));  out = tanh(delta)
// Per-wave: 32-edge tile (two 16-row WMMA M-tiles sharing B fragments).
// GEMMs via v_wmma_f32_16x16x32_f16, fp32 accumulate; tanh via v_tanh_f32.
// B-fragments of each k-step staged in registers to batch global loads.
// ---------------------------------------------------------------------------

typedef __attribute__((ext_vector_type(16))) _Float16 v16h;
typedef __attribute__((ext_vector_type(8)))  _Float16 v8h;
typedef __attribute__((ext_vector_type(4)))  _Float16 v4h;
typedef __attribute__((ext_vector_type(8)))  float    v8f;

__device__ __forceinline__ float fast_tanh(float x) {
#if __has_builtin(__builtin_amdgcn_tanhf)
  return __builtin_amdgcn_tanhf(x);          // v_tanh_f32 (TRANS unit)
#elif __has_builtin(__builtin_amdgcn_tanh_f32)
  return __builtin_amdgcn_tanh_f32(x);
#else
  return tanhf(x);                           // ocml fallback
#endif
}

__device__ __forceinline__ _Float16 h_nan0(float v) {
  if (v != v) v = 0.f;                       // nan_to_num fused with cvt
  return (_Float16)v;
}

__device__ __forceinline__ v4h h4_nan0(float4 v) {
  v4h r;
  r.x = h_nan0(v.x); r.y = h_nan0(v.y); r.z = h_nan0(v.z); r.w = h_nan0(v.w);
  return r;
}

__device__ __forceinline__ v8f wmma_f16(v16h a, v16h b, v8f c) {
  // D = A(16x32 f16) * B(32x16 f16) + C(16x16 f32)
  return __builtin_amdgcn_wmma_f32_16x16x32_f16(
      /*neg_a=*/false, a, /*neg_b=*/false, b,
      /*c_mod=*/(short)0, c, /*reuse_a=*/false, /*reuse_b=*/false);
}

__device__ __forceinline__ v16h frag16(const _Float16* p) {
  // Two contiguous 16B runs: K run0 at p[0..7], K run1 at p[16..23]
  v8h lo = *(const v8h*)p;
  v8h hi = *(const v8h*)(p + 16);
  return __builtin_shufflevector(lo, hi, 0,1,2,3,4,5,6,7,8,9,10,11,12,13,14,15);
}

// A-fragment: 16xK row-major tile (pitch in halves), k-step covers 32 K values.
// Wave32 layout: lane<16 -> M=lane, K = ks*32 + {0..7, 16..23}
//                lane>=16 -> M=lane-16, K = ks*32 + {8..15, 24..31}
__device__ __forceinline__ v16h loadA(const _Float16* base, int pitch, int ks) {
  int lane = threadIdx.x & 31;
  return frag16(base + (lane & 15) * pitch + ks * 32 + ((lane >> 4) << 3));
}

// B-fragment from pre-transposed weights wT[N][Kpad] (K contiguous per column).
__device__ __forceinline__ v16h loadB(const _Float16* wT, int Kpad, int ntile, int ks) {
  int lane = threadIdx.x & 31;
  return frag16(wT + (ntile * 16 + (lane & 15)) * Kpad + ks * 32 + ((lane >> 4) << 3));
}

// One 3-layer MLP on a 32-row tile staged in LDS; scatter-add result.
// X: 32 x (KS1*32) f16 (LDS). H1/H2: 32 x 128 f16 (LDS).
template<int KS1>
__device__ __forceinline__ void run_mlp32(
    const _Float16* X, _Float16* H1, _Float16* H2,
    const _Float16* __restrict__ w1T, const float* __restrict__ b1,
    const _Float16* __restrict__ w2T, const float* __restrict__ b2,
    const _Float16* __restrict__ w3T, const float* __restrict__ b3,
    const int* __restrict__ idx, int e0, int E, float* __restrict__ delta) {
  const int lane = threadIdx.x & 31;
  const int nl   = lane & 15;        // local N (column) within 16x16 D tile
  const int mb   = (lane >> 4) * 8;  // M base within 16-row half: 0 or 8
  constexpr int P1 = KS1 * 32;

  // ---- Layer 1: X(32 x P1) @ W1 -> H1(32 x 128), tanh ----
  {
    v8f acc0[8] = {}, acc1[8] = {};
    for (int k = 0; k < KS1; ++k) {
      v16h a0 = loadA(X,           P1, k);
      v16h a1 = loadA(X + 16 * P1, P1, k);
      v16h bfr[8];
      #pragma unroll
      for (int n = 0; n < 8; ++n) bfr[n] = loadB(w1T, P1, n, k);  // batched
      #pragma unroll
      for (int n = 0; n < 8; ++n) {
        acc0[n] = wmma_f16(a0, bfr[n], acc0[n]);
        acc1[n] = wmma_f16(a1, bfr[n], acc1[n]);
      }
    }
    #pragma unroll
    for (int n = 0; n < 8; ++n) {
      float bias = b1[n * 16 + nl];
      #pragma unroll
      for (int r = 0; r < 8; ++r) {
        H1[(mb + r) * 128 + n * 16 + nl]      = (_Float16)fast_tanh(acc0[n][r] + bias);
        H1[(16 + mb + r) * 128 + n * 16 + nl] = (_Float16)fast_tanh(acc1[n][r] + bias);
      }
    }
    asm volatile("s_wait_dscnt 0" ::: "memory");
  }

  // ---- Layer 2: H1(32x128) @ W2 -> H2(32x128), tanh ----
  {
    v8f acc0[8] = {}, acc1[8] = {};
    for (int k = 0; k < 4; ++k) {
      v16h a0 = loadA(H1,            128, k);
      v16h a1 = loadA(H1 + 16 * 128, 128, k);
      v16h bfr[8];
      #pragma unroll
      for (int n = 0; n < 8; ++n) bfr[n] = loadB(w2T, 128, n, k);
      #pragma unroll
      for (int n = 0; n < 8; ++n) {
        acc0[n] = wmma_f16(a0, bfr[n], acc0[n]);
        acc1[n] = wmma_f16(a1, bfr[n], acc1[n]);
      }
    }
    #pragma unroll
    for (int n = 0; n < 8; ++n) {
      float bias = b2[n * 16 + nl];
      #pragma unroll
      for (int r = 0; r < 8; ++r) {
        H2[(mb + r) * 128 + n * 16 + nl]      = (_Float16)fast_tanh(acc0[n][r] + bias);
        H2[(16 + mb + r) * 128 + n * 16 + nl] = (_Float16)fast_tanh(acc1[n][r] + bias);
      }
    }
    asm volatile("s_wait_dscnt 0" ::: "memory");
  }

  // ---- Layer 3: H2(32x128) @ W3 -> 32x64, tanh, scatter-add ----
  {
    v8f acc0[4] = {}, acc1[4] = {};
    for (int k = 0; k < 4; ++k) {
      v16h a0 = loadA(H2,            128, k);
      v16h a1 = loadA(H2 + 16 * 128, 128, k);
      v16h bfr[4];
      #pragma unroll
      for (int n = 0; n < 4; ++n) bfr[n] = loadB(w3T, 128, n, k);
      #pragma unroll
      for (int n = 0; n < 4; ++n) {
        acc0[n] = wmma_f16(a0, bfr[n], acc0[n]);
        acc1[n] = wmma_f16(a1, bfr[n], acc1[n]);
      }
    }
    int nodes0[8], nodes1[8];
    #pragma unroll
    for (int r = 0; r < 8; ++r) {
      nodes0[r] = idx[min(e0 + mb + r,      E - 1)];
      nodes1[r] = idx[min(e0 + 16 + mb + r, E - 1)];
    }
    #pragma unroll
    for (int n = 0; n < 4; ++n) {
      float bias = b3[n * 16 + nl];
      #pragma unroll
      for (int r = 0; r < 8; ++r) {
        if (e0 + mb + r < E)
          atomicAdd(delta + (size_t)nodes0[r] * 64 + n * 16 + nl,
                    fast_tanh(acc0[n][r] + bias));
        if (e0 + 16 + mb + r < E)
          atomicAdd(delta + (size_t)nodes1[r] * 64 + n * 16 + nl,
                    fast_tanh(acc1[n][r] + bias));
      }
    }
  }
}

// IS_LINE: input = [h[from] (64), h[to] (64), x_line (8), h_g (16), t] pad->160
//          runs lf-MLP (scatter to from) and lt-MLP (scatter to to)
// !IS_LINE: input = [h[bus] (64), x_gen (4), h_g (16), t] pad->96
template<bool IS_LINE>
__global__ __launch_bounds__(64)
void edge_mlp_kernel(
    const float* __restrict__ h_local, const float* __restrict__ h_global,
    const float* __restrict__ x_edge,  const float* __restrict__ t_ptr,
    const int* __restrict__ idxA, const int* __restrict__ idxB,
    const _Float16* __restrict__ w1a, const float* __restrict__ b1a,
    const _Float16* __restrict__ w2a, const float* __restrict__ b2a,
    const _Float16* __restrict__ w3a, const float* __restrict__ b3a,
    const _Float16* __restrict__ w1b, const float* __restrict__ b1b,
    const _Float16* __restrict__ w2b, const float* __restrict__ b2b,
    const _Float16* __restrict__ w3b, const float* __restrict__ b3b,
    float* __restrict__ delta, int E) {
  constexpr int K1 = IS_LINE ? 160 : 96;                 // padded input width
  constexpr int PER_WAVE = K1 * 32 + 128 * 32 * 2;       // halves: X + H1 + H2
  __shared__ __align__(16) _Float16 lds[2 * PER_WAVE];   // 2 waves per block

  const int wave = threadIdx.x >> 5;
  const int lane = threadIdx.x & 31;
  _Float16* X  = lds + wave * PER_WAVE;
  _Float16* H1 = X + K1 * 32;
  _Float16* H2 = H1 + 128 * 32;

  const int numTiles = (E + 31) / 32;
  const int tile = blockIdx.x * 2 + wave;
  if (tile >= numTiles) return;   // waves are fully independent; no block sync
  const int e0 = tile * 32;
  const float tv = t_ptr[0];

  // Gather + fp16-convert the 32-row input tile into LDS (nan_to_num fused).
  // h_local rows: one coalesced float4 (16B) load + one 8B LDS store per lane.
  for (int row = 0; row < 32; ++row) {
    const int e = min(e0 + row, E - 1);   // tail rows duplicate last edge
    if (IS_LINE) {
      const int nf = idxA[e], nt = idxB[e];
      {  // cols 0..127: h[from] | h[to]
        const float* src = (lane < 16)
            ? (h_local + (size_t)nf * 64 + lane * 4)
            : (h_local + (size_t)nt * 64 + (lane - 16) * 4);
        float4 v = *(const float4*)src;
        const int c = (lane < 16) ? lane * 4 : 64 + (lane - 16) * 4;
        *(v4h*)(X + row * 160 + c) = h4_nan0(v);
      }
      {  // cols 128..159: x_line(8) | h_g(16) | t | pad
        const int c = 128 + lane;
        float v;
        if      (c < 136) v = x_edge[(size_t)e * 8 + (c - 128)];
        else if (c < 152) v = h_global[c - 136];
        else if (c == 152) v = tv;
        else               v = 0.f;
        X[row * 160 + c] = h_nan0(v);
      }
    } else {
      const int nb = idxA[e];
      if (lane < 16) {  // cols 0..63: h[bus]
        float4 v = *(const float4*)(h_local + (size_t)nb * 64 + lane * 4);
        *(v4h*)(X + row * 96 + lane * 4) = h4_nan0(v);
      }
      {  // cols 64..95: x_gen(4) | h_g(16) | t | pad
        const int c = 64 + lane;
        float v;
        if      (c < 68) v = x_edge[(size_t)e * 4 + (c - 64)];
        else if (c < 84) v = h_global[c - 68];
        else if (c == 84) v = tv;
        else             v = 0.f;
        X[row * 96 + c] = h_nan0(v);
      }
    }
  }
  asm volatile("s_wait_dscnt 0" ::: "memory");

  run_mlp32<IS_LINE ? 5 : 3>(X, H1, H2, w1a, b1a, w2a, b2a, w3a, b3a,
                             idxA, e0, E, delta);
  if constexpr (IS_LINE)
    run_mlp32<5>(X, H1, H2, w1b, b1b, w2b, b2b, w3b, b3b, idxB, e0, E, delta);
}

// Weights: fp32 [din][dout] row-major -> f16 [dout][Kpad] (K contiguous, 0-pad).
__global__ void prep_weight_kernel(const float* __restrict__ src,
                                   _Float16* __restrict__ dst,
                                   int din, int dout, int Kpad) {
  int i = blockIdx.x * blockDim.x + threadIdx.x;
  if (i >= dout * Kpad) return;
  int n = i / Kpad, k = i % Kpad;
  dst[i] = (_Float16)((k < din) ? src[(size_t)k * dout + n] : 0.f);
}

__global__ void zero4_kernel(float4* __restrict__ p, int n4) {
  int i = blockIdx.x * blockDim.x + threadIdx.x;
  if (i < n4) p[i] = make_float4(0.f, 0.f, 0.f, 0.f);
}

__global__ void tanh_out_kernel(const float4* __restrict__ delta,
                                float4* __restrict__ out, int n4) {
  int i = blockIdx.x * blockDim.x + threadIdx.x;
  if (i < n4) {
    float4 v = delta[i];
    out[i] = make_float4(fast_tanh(v.x), fast_tanh(v.y),
                         fast_tanh(v.z), fast_tanh(v.w));
  }
}

extern "C" void kernel_launch(void* const* d_in, const int* in_sizes, int n_in,
                              void* d_out, int out_size, void* d_ws, size_t ws_size,
                              hipStream_t stream) {
  const float* h_local  = (const float*)d_in[0];
  const float* h_global = (const float*)d_in[1];
  const float* x_line   = (const float*)d_in[2];
  const float* x_gen    = (const float*)d_in[3];
  const float* t_ptr    = (const float*)d_in[4];
  const int*   line_from = (const int*)d_in[5];
  const int*   line_to   = (const int*)d_in[6];
  const int*   gen_bus   = (const int*)d_in[7];

  const int N      = in_sizes[0] / 64;
  const int E_LINE = in_sizes[5];
  const int E_GEN  = in_sizes[7];

  // ---- workspace layout ----
  char* ws = (char*)d_ws;
  float* delta = (float*)ws;
  size_t off = ((size_t)N * 64 * sizeof(float) + 255) & ~(size_t)255;
  auto walloc = [&](size_t halves) {
    _Float16* p = (_Float16*)(ws + off);
    off += (halves * 2 + 255) & ~(size_t)255;
    return p;
  };
  // line MLPs: Kpad 160/128/128 ; gen MLP: Kpad 96/128/128
  _Float16* lf_w1T = walloc(128 * 160);
  _Float16* lf_w2T = walloc(128 * 128);
  _Float16* lf_w3T = walloc(64 * 128);
  _Float16* lt_w1T = walloc(128 * 160);
  _Float16* lt_w2T = walloc(128 * 128);
  _Float16* lt_w3T = walloc(64 * 128);
  _Float16* gb_w1T = walloc(128 * 96);
  _Float16* gb_w2T = walloc(128 * 128);
  _Float16* gb_w3T = walloc(64 * 128);
  (void)ws_size; (void)n_in;

  // ---- weight prep (tiny) ----
  auto prep = [&](const void* src, _Float16* dst, int din, int dout, int Kpad) {
    int n = dout * Kpad;
    prep_weight_kernel<<<(n + 255) / 256, 256, 0, stream>>>(
        (const float*)src, dst, din, dout, Kpad);
  };
  prep(d_in[8],  lf_w1T, 153, 128, 160);
  prep(d_in[10], lf_w2T, 128, 128, 128);
  prep(d_in[12], lf_w3T, 128,  64, 128);
  prep(d_in[14], lt_w1T, 153, 128, 160);
  prep(d_in[16], lt_w2T, 128, 128, 128);
  prep(d_in[18], lt_w3T, 128,  64, 128);
  prep(d_in[20], gb_w1T,  85, 128,  96);
  prep(d_in[22], gb_w2T, 128, 128, 128);
  prep(d_in[24], gb_w3T, 128,  64, 128);

  // ---- zero delta ----
  {
    int n4 = N * 64 / 4;
    zero4_kernel<<<(n4 + 255) / 256, 256, 0, stream>>>((float4*)delta, n4);
  }

  // ---- line edges: lf + lt MLPs, fused gather ----
  {
    int tiles = (E_LINE + 31) / 32;
    edge_mlp_kernel<true><<<(tiles + 1) / 2, 64, 0, stream>>>(
        h_local, h_global, x_line, t_ptr, line_from, line_to,
        lf_w1T, (const float*)d_in[9],  lf_w2T, (const float*)d_in[11],
        lf_w3T, (const float*)d_in[13],
        lt_w1T, (const float*)d_in[15], lt_w2T, (const float*)d_in[17],
        lt_w3T, (const float*)d_in[19],
        delta, E_LINE);
  }

  // ---- gen edges ----
  {
    int tiles = (E_GEN + 31) / 32;
    edge_mlp_kernel<false><<<(tiles + 1) / 2, 64, 0, stream>>>(
        h_local, h_global, x_gen, t_ptr, gen_bus, gen_bus,
        gb_w1T, (const float*)d_in[21], gb_w2T, (const float*)d_in[23],
        gb_w3T, (const float*)d_in[25],
        gb_w1T, (const float*)d_in[21], gb_w2T, (const float*)d_in[23],
        gb_w3T, (const float*)d_in[25],
        delta, E_GEN);
  }

  // ---- out = tanh(delta) ----
  {
    int n4 = N * 64 / 4;
    tanh_out_kernel<<<(n4 + 255) / 256, 256, 0, stream>>>(
        (const float4*)delta, (float4*)d_out, n4);
  }
}
